// BiLSTM_CRF_78202764526048
// MI455X (gfx1250) — matrix operand order repaired and checked by reading, not compiled, not run on hardware
//
#include <hip/hip_runtime.h>
#include <hip/hip_bf16.h>

// ---------------------------------------------------------------------------
// BiLSTM-CRF forward loss for MI455X (gfx1250), wave32 + WMMA.
//
// Decomposition:
//   1) e = emb[X]  -> f16                                  (gather kernel)
//   2) gih_d = e @ Wih_dᵀ + (bih_d + bhh_d)                (big WMMA GEMM ×2)
//   3) 32 recurrent steps per direction:
//        g = gih_d[t] + h @ Whh_dᵀ   (WMMA GEMM [512,128]x[128,512])
//        gates -> h,c  (VALU kernel), h stored f16 for next step + lstm_out
//   4) feats = lstm_out @ fcWᵀ + fcb  (WMMA GEMM, T padded to 64)
//   5) CRF forward algorithm + gold score in one persistent 704-thread block
//
// GEMM: block tile 128x64, 8 waves in 4x2 grid, each wave owns a 32x32
// macro-tile = 2x2 WMMA tiles -> 4 v_wmma per K-chunk per wave, amortizing
// LDS fragment loads (2 A-frags + 2 B-frags feed 4 WMMAs).
// ---------------------------------------------------------------------------

typedef __attribute__((ext_vector_type(16))) _Float16 v16h;
typedef __attribute__((ext_vector_type(8)))  float    v8f;

#define CB 32      // batch (scan/time axis of the LSTM)
#define CS 512     // sequence positions (LSTM "batch")
#define CE 256     // embedding dim
#define CH2 128    // hidden per direction
#define CG 512     // 4*H2
#define CH 256     // hidden both directions
#define CT 22      // tagset + 2
#define CTP 64     // padded tag dim for GEMM
#define CM (CB*CS) // 16384 tokens

// ---------------------------------------------------------------------------
// WMMA GEMM:  C[M,N] = A[M,K] * B[N,K]^T (+bias[N])
// A,B row-major f16, C row-major f32.  M %128==0, N %64==0, K %32==0.
// ---------------------------------------------------------------------------
__global__ __launch_bounds__(256)
void k_gemm_f16(const _Float16* __restrict__ A,
                const _Float16* __restrict__ Bw,
                float* __restrict__ C,
                int M, int N, int K, int ldc,
                const float* __restrict__ bias)
{
    __shared__ _Float16 As[128 * 34];   // 32 K-halves + 2 pad per row
    __shared__ _Float16 Bs[64 * 34];

    const int tid  = threadIdx.x;
    const int lane = tid & 31;
    const int wave = tid >> 5;          // 0..7
    const int bm   = blockIdx.y * 128;
    const int bn   = blockIdx.x * 64;
    const int wm   = (wave & 3) * 32;   // wave m offset within block tile
    const int wn   = (wave >> 2) * 32;  // wave n offset within block tile

    v8f acc00 = {}, acc01 = {}, acc10 = {}, acc11 = {};

    // A tile loader: 128x32 halves, 16 contiguous halves per thread
    const int arow = tid >> 1;          // 0..127
    const int acol = (tid & 1) * 16;    // 0,16
    // B tile loader: 64x32 halves, 8 contiguous halves per thread
    const int brow = tid >> 2;          // 0..63
    const int bcol = (tid & 3) * 8;     // 0,8,16,24

    // fragment index precompute (contiguous k = 16*(lane>>4) + h, same order for A and B)
    const int fm  = lane & 15;
    const int fko = (lane >> 4) * 16;
    const int fn  = lane & 15;
    const int fkb = (lane >> 4) * 16;

    for (int k0 = 0; k0 < K; k0 += 32) {
        const _Float16* ag = A  + (size_t)(bm + arow) * K + k0 + acol;
        const _Float16* bg = Bw + (size_t)(bn + brow) * K + k0 + bcol;
        #pragma unroll
        for (int u = 0; u < 16; ++u)
            As[arow * 34 + acol + u] = ag[u];
        #pragma unroll
        for (int u = 0; u < 8; ++u)
            Bs[brow * 34 + bcol + u] = bg[u];
        __syncthreads();

        // two A fragments (rows wm..wm+15 and wm+16..wm+31)
        v16h af0 = {}, af1 = {};
        #pragma unroll
        for (int h = 0; h < 16; ++h) {
            const int v = h >> 1, p = h & 1;
            const int k = h + fko; (void)v; (void)p;
            af0[h] = As[(wm + fm) * 34 + k];
            af1[h] = As[(wm + 16 + fm) * 34 + k];
        }
        // two B fragments (cols wn..wn+15 and wn+16..wn+31)
        v16h bf0 = {}, bf1 = {};
        #pragma unroll
        for (int h = 0; h < 16; ++h) {
            bf0[h] = Bs[(wn + fn) * 34 + fkb + h];
            bf1[h] = Bs[(wn + 16 + fn) * 34 + fkb + h];
        }

        acc00 = __builtin_amdgcn_wmma_f32_16x16x32_f16(false, af0, false, bf0,
                                                       (short)0, acc00, false, false);
        acc01 = __builtin_amdgcn_wmma_f32_16x16x32_f16(false, af0, false, bf1,
                                                       (short)0, acc01, false, false);
        acc10 = __builtin_amdgcn_wmma_f32_16x16x32_f16(false, af1, false, bf0,
                                                       (short)0, acc10, false, false);
        acc11 = __builtin_amdgcn_wmma_f32_16x16x32_f16(false, af1, false, bf1,
                                                       (short)0, acc11, false, false);
        __syncthreads();
    }

    // C/D layout: VGPR r -> row r (lanes 0-15) / 8+r (lanes 16-31), col = lane&15
    const int row0 = bm + wm + (lane >> 4) * 8;
    const int col0 = bn + wn + (lane & 15);
    const float b0 = bias ? bias[col0]      : 0.f;
    const float b1 = bias ? bias[col0 + 16] : 0.f;
    #pragma unroll
    for (int r = 0; r < 8; ++r) {
        C[(size_t)(row0 + r)      * ldc + col0]      = acc00[r] + b0;
        C[(size_t)(row0 + r)      * ldc + col0 + 16] = acc01[r] + b1;
        C[(size_t)(row0 + 16 + r) * ldc + col0]      = acc10[r] + b0;
        C[(size_t)(row0 + 16 + r) * ldc + col0 + 16] = acc11[r] + b1;
    }
}

// ---------------------------------------------------------------------------
// Embedding gather -> f16.  One block per token, one thread per channel.
// ---------------------------------------------------------------------------
__global__ void k_embed(const int* __restrict__ X, const float* __restrict__ emb,
                        _Float16* __restrict__ e)
{
    const int tokIdx = blockIdx.x;                 // 0..CM-1
    const int c      = threadIdx.x;                // 0..CE-1
    const int tok    = X[tokIdx];
    e[(size_t)tokIdx * CE + c] = (_Float16)emb[(size_t)tok * CE + c];
}

__global__ void k_f32_to_f16(const float* __restrict__ s, _Float16* __restrict__ d, int n)
{
    const int i = blockIdx.x * blockDim.x + threadIdx.x;
    if (i < n) d[i] = (_Float16)s[i];
}

__global__ void k_bias_sum(const float* __restrict__ a, const float* __restrict__ b,
                           float* __restrict__ o, int n)
{
    const int i = blockIdx.x * blockDim.x + threadIdx.x;
    if (i < n) o[i] = a[i] + b[i];
}

// fcW [T,H] -> padded [CTP,H] f16, fcb -> padded [CTP]
__global__ void k_fc_pad(const float* __restrict__ fcW, const float* __restrict__ fcb,
                         _Float16* __restrict__ wdst, float* __restrict__ bdst)
{
    const int i = blockIdx.x * blockDim.x + threadIdx.x;   // CTP*CH
    const int r = i >> 8, c = i & 255;
    wdst[i] = (r < CT) ? (_Float16)fcW[r * CH + c] : (_Float16)0.f;
    if (c == 0) bdst[r] = (r < CT) ? fcb[r] : 0.f;
}

__global__ void k_init_state(const float* __restrict__ h0, const float* __restrict__ c0,
                             _Float16* __restrict__ hF, float* __restrict__ c, int dir)
{
    const int i = blockIdx.x * blockDim.x + threadIdx.x;   // CS*CH2
    hF[i] = (_Float16)h0[dir * (CS * CH2) + i];
    c[i]  = c0[dir * (CS * CH2) + i];
}

// ---------------------------------------------------------------------------
// LSTM gates: g = gbuf (h@Whhᵀ) + gih[t] (already has biases); i,f,g,o split.
// ---------------------------------------------------------------------------
__global__ void k_lstm_gates(const float* __restrict__ gbuf,
                             const float* __restrict__ gih_t,
                             float* __restrict__ c,
                             _Float16* __restrict__ hF16,
                             _Float16* __restrict__ lstmOut,
                             int outRowBase, int dirOff)
{
    const int idx = blockIdx.x * blockDim.x + threadIdx.x; // 0..CS*CH2-1
    const int s = idx >> 7, j = idx & 127;
    const size_t r = (size_t)s * CG;
    const float gi = gbuf[r + j]           + gih_t[r + j];
    const float gf = gbuf[r + CH2 + j]     + gih_t[r + CH2 + j];
    const float gg = gbuf[r + 2 * CH2 + j] + gih_t[r + 2 * CH2 + j];
    const float go = gbuf[r + 3 * CH2 + j] + gih_t[r + 3 * CH2 + j];
    const float si = 1.f / (1.f + __expf(-gi));
    const float sf = 1.f / (1.f + __expf(-gf));
    const float so = 1.f / (1.f + __expf(-go));
    const float c2 = sf * c[idx] + si * tanhf(gg);
    const float h2 = so * tanhf(c2);
    c[idx]    = c2;
    hF16[idx] = (_Float16)h2;
    lstmOut[(size_t)(outRowBase + s) * CH + dirOff + j] = (_Float16)h2;
}

// ---------------------------------------------------------------------------
// CRF: forward algorithm + gold score, one persistent block.
// 704 threads = 32 batches * 22 tags = 22 wave32s.
// feats is [CM, CTP] (cols >= CT are dead padding).
// ---------------------------------------------------------------------------
__global__ __launch_bounds__(704)
void k_crf(const float* __restrict__ feats, const int* __restrict__ Y,
           const float* __restrict__ trans, float* __restrict__ out)
{
    __shared__ float alpha[CB][24];
    __shared__ float trS[CT][CT];
    __shared__ float fwd[CB];
    __shared__ float gold[CB];

    const int t = threadIdx.x;
    const int b = t / CT, n = t % CT;

    if (t < CT * CT) trS[t / CT][t % CT] = trans[t];
    alpha[b][n] = (n == CT - 2) ? 0.f : -10000.f;
    __syncthreads();

    for (int s = 0; s < CS; ++s) {
        float m = -3.0e38f;
        #pragma unroll
        for (int p = 0; p < CT; ++p) {
            const float v = alpha[b][p] + trS[n][p];
            m = (v > m) ? v : m;
        }
        float sum = 0.f;
        #pragma unroll
        for (int p = 0; p < CT; ++p)
            sum += __expf(alpha[b][p] + trS[n][p] - m);
        const float newv = m + __logf(sum) + feats[(size_t)(b * CS + s) * CTP + n];
        __syncthreads();
        alpha[b][n] = newv;
        __syncthreads();
    }

    // terminal logsumexp with trans[T-1, :]
    if (n == 0) {
        float m = -3.0e38f;
        #pragma unroll
        for (int p = 0; p < CT; ++p) {
            const float v = alpha[b][p] + trS[CT - 1][p];
            m = (v > m) ? v : m;
        }
        float sum = 0.f;
        #pragma unroll
        for (int p = 0; p < CT; ++p)
            sum += __expf(alpha[b][p] + trS[CT - 1][p] - m);
        fwd[b] = m + __logf(sum);
    }

    // gold score: one thread per batch element
    if (t < CB) {
        float emit = 0.f, tr = 0.f;
        int prev = CT - 2;                       // start tag
        for (int s = 0; s < CS; ++s) {
            const int cur = Y[t * CS + s];
            emit += feats[(size_t)(t * CS + s) * CTP + cur];
            tr   += trS[cur][prev];
            prev  = cur;
        }
        tr += trS[CT - 1][prev];                 // stop tag
        gold[t] = emit + tr;
    }
    __syncthreads();

    if (t == 0) {
        float fs = 0.f, gs = 0.f;
        for (int bb = 0; bb < CB; ++bb) { fs += fwd[bb]; gs += gold[bb]; }
        out[0] = fs / (float)CB - gs / (float)CB;
    }
}

__global__ void k_copyY(const int* __restrict__ Y, float* __restrict__ out)
{
    const int i = blockIdx.x * blockDim.x + threadIdx.x;
    if (i < CM) out[1 + i] = (float)Y[i];
}

// ---------------------------------------------------------------------------
// Host side
// ---------------------------------------------------------------------------
static inline size_t alignUp(size_t x) { return (x + 255) & ~(size_t)255; }

extern "C" void kernel_launch(void* const* d_in, const int* in_sizes, int n_in,
                              void* d_out, int out_size, void* d_ws, size_t ws_size,
                              hipStream_t stream) {
    (void)in_sizes; (void)n_in; (void)out_size; (void)ws_size;

    const int*   X     = (const int*)d_in[0];
    const int*   Y     = (const int*)d_in[1];
    const float* emb   = (const float*)d_in[2];
    const float* Wih_f = (const float*)d_in[3];
    const float* Whh_f = (const float*)d_in[4];
    const float* bih_f = (const float*)d_in[5];
    const float* bhh_f = (const float*)d_in[6];
    const float* Wih_b = (const float*)d_in[7];
    const float* Whh_b = (const float*)d_in[8];
    const float* bih_b = (const float*)d_in[9];
    const float* bhh_b = (const float*)d_in[10];
    const float* fcW   = (const float*)d_in[11];
    const float* fcb   = (const float*)d_in[12];
    const float* trans = (const float*)d_in[13];
    const float* h0    = (const float*)d_in[14];
    const float* c0    = (const float*)d_in[15];
    float* out = (float*)d_out;

    // workspace carve-up
    char* w = (char*)d_ws;
    size_t off = 0;
    auto take = [&](size_t bytes) { char* p = w + off; off += alignUp(bytes); return p; };

    _Float16* eF16   = (_Float16*)take((size_t)CM * CE * 2);
    _Float16* wihF   = (_Float16*)take((size_t)CG * CE * 2);
    _Float16* wihB   = (_Float16*)take((size_t)CG * CE * 2);
    _Float16* whhF   = (_Float16*)take((size_t)CG * CH2 * 2);
    _Float16* whhB   = (_Float16*)take((size_t)CG * CH2 * 2);
    _Float16* fcwP   = (_Float16*)take((size_t)CTP * CH * 2);
    float*    bsumF  = (float*)take(CG * 4);
    float*    bsumB  = (float*)take(CG * 4);
    float*    fcbP   = (float*)take(CTP * 4);
    float*    gihF   = (float*)take((size_t)CM * CG * 4);
    float*    gihB   = (float*)take((size_t)CM * CG * 4);
    float*    gbuf   = (float*)take((size_t)CS * CG * 4);
    _Float16* hF16   = (_Float16*)take((size_t)CS * CH2 * 2);
    float*    cSt    = (float*)take((size_t)CS * CH2 * 4);
    _Float16* lstmO  = (_Float16*)take((size_t)CM * CH * 2);
    float*    feats  = (float*)take((size_t)CM * CTP * 4);

    // 1) embedding gather -> f16
    k_embed<<<CM, CE, 0, stream>>>(X, emb, eF16);

    // 2) weight conversions / bias folds
    k_f32_to_f16<<<(CG * CE + 255) / 256, 256, 0, stream>>>(Wih_f, wihF, CG * CE);
    k_f32_to_f16<<<(CG * CE + 255) / 256, 256, 0, stream>>>(Wih_b, wihB, CG * CE);
    k_f32_to_f16<<<(CG * CH2 + 255) / 256, 256, 0, stream>>>(Whh_f, whhF, CG * CH2);
    k_f32_to_f16<<<(CG * CH2 + 255) / 256, 256, 0, stream>>>(Whh_b, whhB, CG * CH2);
    k_bias_sum<<<(CG + 255) / 256, 256, 0, stream>>>(bih_f, bhh_f, bsumF, CG);
    k_bias_sum<<<(CG + 255) / 256, 256, 0, stream>>>(bih_b, bhh_b, bsumB, CG);
    k_fc_pad<<<(CTP * CH + 255) / 256, 256, 0, stream>>>(fcW, fcb, fcwP, fcbP);

    // 3) input projections for both directions: gih = e @ Wihᵀ + (bih+bhh)
    {
        dim3 grid(CG / 64, CM / 128);
        k_gemm_f16<<<grid, 256, 0, stream>>>(eF16, wihF, gihF, CM, CG, CE, CG, bsumF);
        k_gemm_f16<<<grid, 256, 0, stream>>>(eF16, wihB, gihB, CM, CG, CE, CG, bsumB);
    }

    // 4) recurrent scans (time axis = B = 32 steps), both directions
    for (int dir = 0; dir < 2; ++dir) {
        const _Float16* whh  = dir ? whhB : whhF;
        const float*    gih  = dir ? gihB : gihF;
        k_init_state<<<(CS * CH2) / 256, 256, 0, stream>>>(h0, c0, hF16, cSt, dir);
        for (int t = 0; t < CB; ++t) {
            const int tIdx = dir ? (CB - 1 - t) : t;
            dim3 grid(CG / 64, CS / 128);
            k_gemm_f16<<<grid, 256, 0, stream>>>(hF16, whh, gbuf, CS, CG, CH2, CG, nullptr);
            k_lstm_gates<<<(CS * CH2) / 256, 256, 0, stream>>>(
                gbuf, gih + (size_t)tIdx * CS * CG, cSt, hF16, lstmO,
                tIdx * CS, dir * CH2);
        }
    }

    // 5) FC: feats = lstm_out @ fcWᵀ + fcb  (T padded to 64)
    {
        dim3 grid(CTP / 64, CM / 128);
        k_gemm_f16<<<grid, 256, 0, stream>>>(lstmO, fcwP, feats, CM, CTP, CH, CTP, fcbP);
    }

    // 6) CRF loss -> out[0]
    k_crf<<<1, 704, 0, stream>>>(feats, Y, trans, out);

    // 7) second tuple element: Y
    k_copyY<<<(CM + 255) / 256, 256, 0, stream>>>(Y, out);
}
